// EncoderDrugPyG_57260503990643
// MI455X (gfx1250) — matrix operand order repaired and checked by reading, not compiled, not run on hardware
//
#include <hip/hip_runtime.h>
#include <hip/hip_bf16.h>

// ---------------------------------------------------------------------------
// Problem constants (fixed by the reference)
// ---------------------------------------------------------------------------
#define NNODES 131072
#define NEDGES 393216
#define DF     128
#define LNEPS  1e-5f

// ---------------------------------------------------------------------------
// WMMA plumbing: prefer exact-precision f32 16x16x4 WMMA; fall back to the
// codegen-confirmed bf16 16x16x32 WMMA if the f32 builtin is not declared.
// ---------------------------------------------------------------------------
typedef float  v2f   __attribute__((ext_vector_type(2)));
typedef float  v8f   __attribute__((ext_vector_type(8)));
typedef __bf16 v16bf __attribute__((ext_vector_type(16)));

#if defined(__has_builtin)
#if __has_builtin(__builtin_amdgcn_wmma_f32_16x16x4_f32)
#define WMMA_F32_NATIVE 1
#endif
#endif
#ifndef WMMA_F32_NATIVE
#define WMMA_F32_NATIVE 0
#endif

// ---------------------------------------------------------------------------
// Degree / normalization kernels:  dinv[i] = (1 + indeg(i))^-0.5
// ---------------------------------------------------------------------------
__global__ void __launch_bounds__(256)
gcn_deg_init(float* __restrict__ dinv) {
    dinv[blockIdx.x * 256 + threadIdx.x] = 1.0f;   // self-loop contribution
}

__global__ void __launch_bounds__(256)
gcn_deg_accum(float* __restrict__ dinv, const long long* __restrict__ ei) {
    int e = blockIdx.x * 256 + threadIdx.x;
    if (e < NEDGES) {
        long long d = ei[NEDGES + e];              // dst row of edge_index
        atomicAdd(&dinv[d], 1.0f);
    }
}

__global__ void __launch_bounds__(256)
gcn_deg_finalize(float* __restrict__ dinv) {
    int i = blockIdx.x * 256 + threadIdx.x;
    dinv[i] = rsqrtf(dinv[i]);                     // deg >= 1 always
}

// ---------------------------------------------------------------------------
// Tiled GEMM  Out[N x 128] = relu?(Ain)[N x 128] @ W[128 x 128]   via WMMA.
// Block: 256 threads = 8 waves. Block covers 16 rows; wave w covers cols
// [16w, 16w+16). W staged in LDS in two 32KB K-halves; A tile staged with
// pad-129 stride (bank-conflict free A-fragment reads). ReLU fused on load.
// ---------------------------------------------------------------------------
__global__ void __launch_bounds__(256)
gcn_gemm_wmma(const float* __restrict__ Ain, const float* __restrict__ W,
              float* __restrict__ Out, int applyRelu)
{
    __shared__ float sW[64 * DF];     // 32 KB: one K-half of W
    __shared__ float sA[16 * 129];    // ~8 KB: padded 16x128 A tile

    const int tid  = threadIdx.x;
    const int wave = tid >> 5;
    const int lane = tid & 31;
    const int half = lane >> 4;       // 0 = lanes 0-15, 1 = lanes 16-31
    const int ml   = lane & 15;
    const int row0 = blockIdx.x * 16;
    const int col0 = wave * 16;

    // Stage A tile with fused ReLU (coalesced in k).
    for (int i = tid; i < 16 * DF; i += 256) {
        const int m = i >> 7, k = i & (DF - 1);
        float v = Ain[(size_t)(row0 + m) * DF + k];
        if (applyRelu) v = fmaxf(v, 0.0f);
        sA[m * 129 + k] = v;
    }

    v8f acc = {0.f, 0.f, 0.f, 0.f, 0.f, 0.f, 0.f, 0.f};

    for (int kh = 0; kh < DF; kh += 64) {
        __syncthreads();              // protect sW against overwrite
        {   // stage 64 K-rows of W (float4 coalesced)
            const float4* Wh  = (const float4*)(W + (size_t)kh * DF);
            float4*       sW4 = (float4*)sW;
            for (int i = tid; i < 64 * DF / 4; i += 256) sW4[i] = Wh[i];
        }
        __syncthreads();

#if WMMA_F32_NATIVE
        // V_WMMA_F32_16X16X4_F32: A 16x4 (2 VGPR), B 4x16 (2 VGPR), C 16x16.
        // A: v[0] = A[m][k0 + 2*half], v[1] = A[m][k0 + 2*half + 1]
        // B: v[0] = B[k0 + 2*half][n], v[1] = B[k0 + 2*half + 1][n]
        for (int k0 = 0; k0 < 64; k0 += 4) {
            v2f a, b;
            const int kA = kh + k0 + half * 2;
            a.x = sA[ml * 129 + kA];
            a.y = sA[ml * 129 + kA + 1];
            const int kB = k0 + half * 2;
            b.x = sW[kB * DF + col0 + ml];
            b.y = sW[(kB + 1) * DF + col0 + ml];
            acc = __builtin_amdgcn_wmma_f32_16x16x4_f32(
                false, a, false, b, (short)0, acc, false, false);
        }
#else
        // Fallback: V_WMMA_F32_16X16X32_BF16 (f32 accumulate).
        for (int k0 = 0; k0 < 64; k0 += 32) {
            v16bf a, b;
            const int baseA = kh + k0 + half * 16;
            for (int j = 0; j < 8; ++j) {
                a[j]     = (__bf16)sA[ml * 129 + baseA + j];
                a[j + 8] = (__bf16)sA[ml * 129 + baseA + 8 + j];
            }
            const int baseB = k0 + half * 16;
            for (int j = 0; j < 16; ++j)
                b[j] = (__bf16)sW[(baseB + j) * DF + col0 + ml];
            acc = __builtin_amdgcn_wmma_f32_16x16x32_bf16(
                false, a, false, b, (short)0, acc, false, false);
        }
#endif
    }

    // C/D layout: VGPR r -> (M = r + 8*half, N = col0 + ml)
    for (int r = 0; r < 8; ++r)
        Out[(size_t)(row0 + r + half * 8) * DF + col0 + ml] = acc[r];
}

// ---------------------------------------------------------------------------
// Out[i][:] = bias[:] + dinv[i]^2 * T[i][:]       (bias + self-loop message)
// One thread per float4.
// ---------------------------------------------------------------------------
__global__ void __launch_bounds__(256)
gcn_bias_selfloop(const float* __restrict__ T, const float* __restrict__ bias,
                  const float* __restrict__ dinv, float* __restrict__ Out)
{
    const size_t gid = (size_t)blockIdx.x * 256 + threadIdx.x;  // float4 index
    const int row = (int)(gid >> 5);
    const int j4  = (int)(gid & 31);
    const float di = dinv[row];
    const float c  = di * di;
    const float4 t = ((const float4*)T)[gid];
    const float4 b = ((const float4*)bias)[j4];
    float4 o;
    o.x = b.x + c * t.x;  o.y = b.y + c * t.y;
    o.z = b.z + c * t.z;  o.w = b.w + c * t.w;
    ((float4*)Out)[gid] = o;
}

// ---------------------------------------------------------------------------
// Edge scatter: one wave per edge, lane l owns features [4l, 4l+4).
// Out[dst][:] += dinv[src]*dinv[dst] * T[src][:]   via f32 global atomics.
// ---------------------------------------------------------------------------
__global__ void __launch_bounds__(256)
gcn_edge_scatter(const float* __restrict__ T, const float* __restrict__ dinv,
                 const long long* __restrict__ ei, float* __restrict__ Out)
{
    const int e    = (int)(((unsigned)(blockIdx.x * 256 + threadIdx.x)) >> 5);
    const int lane = threadIdx.x & 31;
    if (e >= NEDGES) return;
    const long long s = ei[e];
    const long long d = ei[NEDGES + e];
    const float coef  = dinv[s] * dinv[d];
    const float4 v    = ((const float4*)(T + (size_t)s * DF))[lane];
    float* dst = Out + (size_t)d * DF + lane * 4;
    atomicAdd(dst + 0, v.x * coef);
    atomicAdd(dst + 1, v.y * coef);
    atomicAdd(dst + 2, v.z * coef);
    atomicAdd(dst + 3, v.w * coef);
}

// ---------------------------------------------------------------------------
// Fused ReLU + LayerNorm + write tokens. batch is contiguous with exactly 32
// nodes/graph, so tokens == layernorm(relu(h)) reshaped — a direct row write.
// One wave per row; wave32 shuffle reduction.
// ---------------------------------------------------------------------------
__global__ void __launch_bounds__(256)
gcn_layernorm_out(const float* __restrict__ H, const float* __restrict__ g,
                  const float* __restrict__ beta, float* __restrict__ Out)
{
    const int wave = threadIdx.x >> 5;
    const int lane = threadIdx.x & 31;
    const int row  = blockIdx.x * 8 + wave;

    const float4 hv = ((const float4*)(H + (size_t)row * DF))[lane];
    float4 v;
    v.x = fmaxf(hv.x, 0.f); v.y = fmaxf(hv.y, 0.f);
    v.z = fmaxf(hv.z, 0.f); v.w = fmaxf(hv.w, 0.f);

    float s  = v.x + v.y + v.z + v.w;
    float sq = v.x * v.x + v.y * v.y + v.z * v.z + v.w * v.w;
    #pragma unroll
    for (int off = 16; off >= 1; off >>= 1) {
        s  += __shfl_xor(s,  off, 32);
        sq += __shfl_xor(sq, off, 32);
    }
    const float mu  = s * (1.0f / DF);
    const float var = sq * (1.0f / DF) - mu * mu;
    const float rs  = rsqrtf(var + LNEPS);

    const float4 gv = ((const float4*)g)[lane];
    const float4 bv = ((const float4*)beta)[lane];
    float4 o;
    o.x = (v.x - mu) * rs * gv.x + bv.x;
    o.y = (v.y - mu) * rs * gv.y + bv.y;
    o.z = (v.z - mu) * rs * gv.z + bv.z;
    o.w = (v.w - mu) * rs * gv.w + bv.w;
    ((float4*)(Out + (size_t)row * DF))[lane] = o;
}

// ---------------------------------------------------------------------------
// Mask tail: every (graph, pos) slot is filled -> mask is all False (0).
// ---------------------------------------------------------------------------
__global__ void __launch_bounds__(256)
gcn_mask_zero(float* __restrict__ p, int n) {
    int i = blockIdx.x * 256 + threadIdx.x;
    if (i < n) p[i] = 0.0f;
}

// ---------------------------------------------------------------------------
// Launch glue
// ---------------------------------------------------------------------------
extern "C" void kernel_launch(void* const* d_in, const int* in_sizes, int n_in,
                              void* d_out, int out_size, void* d_ws, size_t ws_size,
                              hipStream_t stream) {
    (void)in_sizes; (void)n_in; (void)ws_size;
    const float* x    = (const float*)d_in[0];
    const float* Ws[3] = { (const float*)d_in[1], (const float*)d_in[3], (const float*)d_in[5] };
    const float* bs[3] = { (const float*)d_in[2], (const float*)d_in[4], (const float*)d_in[6] };
    const float* ln_g = (const float*)d_in[7];
    const float* ln_b = (const float*)d_in[8];
    const long long* ei = (const long long*)d_in[9];
    // d_in[10] (batch) is structurally i/32 -> unused.

    float* out  = (float*)d_out;
    char*  ws   = (char*)d_ws;
    float* dinv = (float*)ws;                                          // 512 KB
    float* bufT = (float*)(ws + (1u << 20));                           // 64 MB (tmp = h@W)
    float* bufA = (float*)(ws + (1u << 20) + (size_t)NNODES * DF * 4); // 64 MB (h)

    // Normalization coefficients.
    gcn_deg_init    <<<NNODES / 256, 256, 0, stream>>>(dinv);
    gcn_deg_accum   <<<NEDGES / 256, 256, 0, stream>>>(dinv, ei);
    gcn_deg_finalize<<<NNODES / 256, 256, 0, stream>>>(dinv);

    // Three GCN layers (ReLU fused into the next consumer's load).
    const float* h = x;
    for (int l = 0; l < 3; ++l) {
        gcn_gemm_wmma    <<<NNODES / 16, 256, 0, stream>>>(h, Ws[l], bufT, l > 0);
        gcn_bias_selfloop<<<(NNODES * 32) / 256, 256, 0, stream>>>(bufT, bs[l], dinv, bufA);
        gcn_edge_scatter <<<(NEDGES * 32) / 256, 256, 0, stream>>>(bufT, dinv, ei, bufA);
        h = bufA;
    }

    // ReLU + LayerNorm + token write, then all-False mask tail.
    gcn_layernorm_out<<<NNODES / 8, 256, 0, stream>>>(bufA, ln_g, ln_b, out);
    const int maskElems = out_size - NNODES * DF;
    if (maskElems > 0) {
        gcn_mask_zero<<<(maskElems + 255) / 256, 256, 0, stream>>>(
            out + (size_t)NNODES * DF, maskElems);
    }
}
